// CrossAttenHead_56083682951733
// MI455X (gfx1250) — matrix operand order repaired and checked by reading, not compiled, not run on hardware
//
#include <hip/hip_runtime.h>

// ---------------------------------------------------------------------------
// CDNA5 (gfx1250) implementation of CrossAttenHead forward.
// Matrix work runs on v_wmma_f32_16x16x32_bf16 (wave32, f32 accumulate).
// Edge handling in GEMM uses clamp+mask-multiply (branch-free inner loop).
// ---------------------------------------------------------------------------

typedef __attribute__((ext_vector_type(16))) __bf16 v16bf;
typedef __attribute__((ext_vector_type(8)))  float  v8f;

#define BATCH 8
#define NPQ   100      // proposals
#define HIDC  256      // hidden
#define NHEAD 8
#define HDIM  32
#define HWSZ  16384    // 128*128
#define FFC_  2048
#define NW    103      // NP + KS
#define KSEG  8        // K-split for pooling GEMM
#define EPSF  1e-5f

// ---------------------------------------------------------------------------
// Generic batched GEMM: Y[b,m,n] = act( sum_k A[b,m,k] * W[b,k,n] + bias[n] )
// A: f32, row stride lda, batch stride sAb.   W: f32 (K,N) row-major,
// batch stride sWb (0 => shared weights).  One wave32 per 16x16 tile.
// Main K loop is branch-free: OOB rows/cols are clamped to a valid address
// and zeroed by multiplication, so loads issue unconditionally (no exec
// toggling, no per-load waits).  K tail (K%32) runs one masked iteration.
// ---------------------------------------------------------------------------
__global__ void __launch_bounds__(32)
gemm_bf16_kernel(const float* __restrict__ A, long long sAb, int lda,
                 const float* __restrict__ W, long long sWb,
                 const float* __restrict__ bias,
                 float* __restrict__ Y, long long sYb,
                 int M, int N, int K, int relu)
{
    const int lane = threadIdx.x;       // 0..31
    const int hi   = lane >> 4;         // half-wave select
    const int lo   = lane & 15;
    const int n0   = blockIdx.x * 16;
    const int m0   = blockIdx.y * 16;
    const int b    = blockIdx.z;

    const int   mrow   = m0 + lo;
    const int   ncol   = n0 + lo;
    const float mmask  = (mrow < M) ? 1.f : 0.f;
    const float nmask  = (ncol < N) ? 1.f : 0.f;
    const int   mclamp = (mrow < M) ? mrow : (M - 1);
    const int   nclamp = (ncol < N) ? ncol : (N - 1);

    // per-lane base pointers (in-bounds by construction)
    const float* __restrict__ arow = A + (long long)b * sAb
                                       + (long long)mclamp * lda + 8 * hi;
    const float* __restrict__ wcol = W + (long long)b * sWb + nclamp;
    const long long nstride = N;

    v8f acc = {};
    const int kmain = K & ~31;
    int k0 = 0;
    for (; k0 < kmain; k0 += 32) {
        // A fragment (16-bit A 16x32 layout): K chunks {k0+8hi..+7},{+16..}
        v16bf av;
        #pragma unroll
        for (int i = 0; i < 8; ++i) {
            av[i]     = (__bf16)(arow[k0 + i]      * mmask);
            av[i + 8] = (__bf16)(arow[k0 + 16 + i] * mmask);
        }
        // B fragment: fixed column, K chunks {k0+8hi..+7},{+16..} (matches A)
        v16bf bv;
        const long long kb = k0 + 8 * hi;
        #pragma unroll
        for (int e = 0; e < 16; ++e)
            bv[e] = (__bf16)(wcol[(kb + e + (e >= 8 ? 8 : 0)) * nstride] * nmask);

        acc = __builtin_amdgcn_wmma_f32_16x16x32_bf16(false, av, false, bv,
                                                      (short)0, acc, false, false);
    }
    if (k0 < K) {   // tail: one masked iteration (branch-free clamps)
        v16bf av;
        #pragma unroll
        for (int i = 0; i < 8; ++i) {
            const int k1 = k0 + i;
            const int k2 = k0 + 16 + i;
            const int k1c = (k1 + 8 * hi < K) ? k1 : 0;
            const int k2c = (k2 + 8 * hi < K) ? k2 : 0;
            const float g1 = (k1 + 8 * hi < K) ? mmask : 0.f;
            const float g2 = (k2 + 8 * hi < K) ? mmask : 0.f;
            av[i]     = (__bf16)(arow[k1c] * g1);
            av[i + 8] = (__bf16)(arow[k2c] * g2);
        }
        v16bf bv;
        #pragma unroll
        for (int e = 0; e < 16; ++e) {
            const int kk = k0 + 8 * hi + e + (e >= 8 ? 8 : 0);
            const int kc = (kk < K) ? kk : 0;
            const float g = (kk < K) ? nmask : 0.f;
            bv[e] = (__bf16)(wcol[(long long)kc * nstride] * g);
        }
        acc = __builtin_amdgcn_wmma_f32_16x16x32_bf16(false, av, false, bv,
                                                      (short)0, acc, false, false);
    }
    // D store: VGPR r -> row m0 + r + 8*hi, col n0 + lo
    #pragma unroll
    for (int r = 0; r < 8; ++r) {
        const int m = m0 + r + 8 * hi;
        const int n = n0 + lo;
        if (m < M && n < N) {
            float y = acc[r];
            if (bias) y += bias[n];
            if (relu) y = fmaxf(y, 0.f);
            Y[(long long)b * sYb + (long long)m * N + n] = y;
        }
    }
}

// ---------------------------------------------------------------------------
// Pooling GEMM: f[b,n,c] = sum_hw [mask_preds>0] * features[b,c,hw]
// (sigmoid(x)>0.5 <=> x>0).  K split 8 ways, combined with f32 global atomics.
// ---------------------------------------------------------------------------
__global__ void __launch_bounds__(32)
pool_gemm_kernel(const float* __restrict__ mp,
                 const float* __restrict__ feat,
                 float* __restrict__ fout)
{
    const int lane = threadIdx.x;
    const int hi   = lane >> 4;
    const int lo   = lane & 15;
    const int n0   = blockIdx.x * 16;          // C tile
    const int m0   = blockIdx.y * 16;          // NP tile
    const int b    = blockIdx.z / KSEG;
    const int seg  = blockIdx.z % KSEG;
    const int kbeg = seg * (HWSZ / KSEG);
    const int kend = kbeg + (HWSZ / KSEG);

    const int  mrow = m0 + lo;
    const bool mok  = (mrow < NPQ);
    const float one = mok ? 1.f : 0.f;
    const float* arow = mp   + (long long)(b * NPQ  + (mok ? mrow : 0)) * HWSZ;
    const float* brow = feat + (long long)(b * HIDC + n0 + lo) * HWSZ;

    v8f acc = {};
    for (int k0 = kbeg; k0 < kend; k0 += 32) {
        const int ka = k0 + 8 * hi;
        const float4 a0 = *(const float4*)(arow + ka);
        const float4 a1 = *(const float4*)(arow + ka + 4);
        const float4 a2 = *(const float4*)(arow + ka + 16);
        const float4 a3 = *(const float4*)(arow + ka + 20);
        v16bf av;
        #define BIN_(x) ((__bf16)(((x) > 0.f) ? one : 0.f))
        av[0]=BIN_(a0.x);  av[1]=BIN_(a0.y);  av[2]=BIN_(a0.z);  av[3]=BIN_(a0.w);
        av[4]=BIN_(a1.x);  av[5]=BIN_(a1.y);  av[6]=BIN_(a1.z);  av[7]=BIN_(a1.w);
        av[8]=BIN_(a2.x);  av[9]=BIN_(a2.y);  av[10]=BIN_(a2.z); av[11]=BIN_(a2.w);
        av[12]=BIN_(a3.x); av[13]=BIN_(a3.y); av[14]=BIN_(a3.z); av[15]=BIN_(a3.w);
        #undef BIN_

        const int kb = k0 + 8 * hi;
        const float4 b0 = *(const float4*)(brow + kb);
        const float4 b1 = *(const float4*)(brow + kb + 4);
        const float4 b2 = *(const float4*)(brow + kb + 16);
        const float4 b3 = *(const float4*)(brow + kb + 20);
        v16bf bv;
        bv[0]=(__bf16)b0.x;  bv[1]=(__bf16)b0.y;  bv[2]=(__bf16)b0.z;  bv[3]=(__bf16)b0.w;
        bv[4]=(__bf16)b1.x;  bv[5]=(__bf16)b1.y;  bv[6]=(__bf16)b1.z;  bv[7]=(__bf16)b1.w;
        bv[8]=(__bf16)b2.x;  bv[9]=(__bf16)b2.y;  bv[10]=(__bf16)b2.z; bv[11]=(__bf16)b2.w;
        bv[12]=(__bf16)b3.x; bv[13]=(__bf16)b3.y; bv[14]=(__bf16)b3.z; bv[15]=(__bf16)b3.w;

        __builtin_prefetch(brow + kb + 512, 0, 0);   // global_prefetch_b8
        __builtin_prefetch(arow + ka + 512, 0, 0);

        acc = __builtin_amdgcn_wmma_f32_16x16x32_bf16(false, av, false, bv,
                                                      (short)0, acc, false, false);
    }
    #pragma unroll
    for (int r = 0; r < 8; ++r) {
        const int m = m0 + r + 8 * hi;
        if (m < NPQ)
            unsafeAtomicAdd(&fout[(long long)(b * NPQ + m) * HIDC + n0 + lo], acc[r]);
    }
}

// ---------------------------------------------------------------------------
// LayerNorm over last dim (256), optional residual input and relu.
// One 256-thread block per row.
// ---------------------------------------------------------------------------
__global__ void ln_kernel(const float* __restrict__ x, const float* __restrict__ resid,
                          const float* __restrict__ w, const float* __restrict__ bb,
                          float* __restrict__ y, int relu)
{
    __shared__ float red[HIDC];
    const int row = blockIdx.x;
    const int c   = threadIdx.x;
    float v = x[(long long)row * HIDC + c];
    if (resid) v += resid[(long long)row * HIDC + c];

    red[c] = v; __syncthreads();
    for (int s = HIDC / 2; s > 0; s >>= 1) { if (c < s) red[c] += red[c + s]; __syncthreads(); }
    const float mu = red[0] * (1.f / HIDC);
    __syncthreads();

    const float d = v - mu;
    red[c] = d * d; __syncthreads();
    for (int s = HIDC / 2; s > 0; s >>= 1) { if (c < s) red[c] += red[c + s]; __syncthreads(); }
    const float var = red[0] * (1.f / HIDC);

    float out = d * rsqrtf(var + EPSF) * w[c] + bb[c];
    if (relu) out = fmaxf(out, 0.f);
    y[(long long)row * HIDC + c] = out;
}

// ---------------------------------------------------------------------------
// Dynamic depthwise 3-tap conv along channels: tmp[c] = relu(v[c-1]*dw0 +
// v[c]*dw1 + v[c+1]*dw2), zero padded.  One block per (b,n) row.
// ---------------------------------------------------------------------------
__global__ void dysep_conv_kernel(const float* __restrict__ wbuf,
                                  const float* __restrict__ v,
                                  float* __restrict__ out)
{
    __shared__ float sv[HIDC];
    const int row = blockIdx.x;
    const int c   = threadIdx.x;
    sv[c] = v[(long long)row * HIDC + c];
    __syncthreads();
    const float dw0 = wbuf[(long long)row * NW + 0];
    const float dw1 = wbuf[(long long)row * NW + 1];
    const float dw2 = wbuf[(long long)row * NW + 2];
    const float l = (c > 0)        ? sv[c - 1] : 0.f;
    const float r = (c < HIDC - 1) ? sv[c + 1] : 0.f;
    out[(long long)row * HIDC + c] = fmaxf(l * dw0 + sv[c] * dw1 + r * dw2, 0.f);
}

// ---------------------------------------------------------------------------
// Multi-head attention core (100x100 per head is tiny -> LDS/VALU kernel).
// qkv: (B, NP, 768).  o: (B, NP, 256).  One 128-thread block per (b,h).
// ---------------------------------------------------------------------------
__global__ void mha_kernel(const float* __restrict__ qkv, float* __restrict__ o)
{
    const int b   = blockIdx.x / NHEAD;
    const int h   = blockIdx.x % NHEAD;
    const int tid = threadIdx.x;           // 128
    __shared__ float sK[NPQ * HDIM], sV[NPQ * HDIM], sQ[HDIM], sc[NPQ], red[128];

    const float* base = qkv + (long long)b * NPQ * (3 * HIDC);
    for (int i = tid; i < NPQ * HDIM; i += 128) {
        const int t = i / HDIM, d = i % HDIM;
        sK[i] = base[t * (3 * HIDC) + HIDC     + h * HDIM + d];
        sV[i] = base[t * (3 * HIDC) + 2 * HIDC + h * HDIM + d];
    }
    __syncthreads();

    for (int qi = 0; qi < NPQ; ++qi) {
        if (tid < HDIM) sQ[tid] = base[qi * (3 * HIDC) + h * HDIM + tid];
        __syncthreads();
        float s = 0.f;
        if (tid < NPQ) {
            #pragma unroll 4
            for (int d = 0; d < HDIM; ++d) s += sQ[d] * sK[tid * HDIM + d];
            s *= 0.17677669529663687f;     // 1/sqrt(32)
        }
        red[tid] = (tid < NPQ) ? s : -1e30f;
        __syncthreads();
        for (int st = 64; st > 0; st >>= 1) {
            if (tid < st) red[tid] = fmaxf(red[tid], red[tid + st]);
            __syncthreads();
        }
        const float mx = red[0]; __syncthreads();
        const float e = (tid < NPQ) ? __expf(s - mx) : 0.f;
        red[tid] = e; __syncthreads();
        for (int st = 64; st > 0; st >>= 1) {
            if (tid < st) red[tid] += red[tid + st];
            __syncthreads();
        }
        const float inv = 1.f / red[0]; __syncthreads();
        if (tid < NPQ) sc[tid] = e * inv;
        __syncthreads();
        if (tid < HDIM) {
            float acc = 0.f;
            for (int t = 0; t < NPQ; ++t) acc += sc[t] * sV[t * HDIM + tid];
            o[((long long)b * NPQ + qi) * HIDC + h * HDIM + tid] = acc;
        }
        __syncthreads();
    }
}

__global__ void zero_kernel(float* p, int n)
{
    const int i = blockIdx.x * blockDim.x + threadIdx.x;
    if (i < n) p[i] = 0.f;
}

__global__ void copy_kernel(const float* __restrict__ s, float* __restrict__ d, int n)
{
    const int i = blockIdx.x * blockDim.x + threadIdx.x;
    if (i < n) d[i] = s[i];
}

// ---------------------------------------------------------------------------
extern "C" void kernel_launch(void* const* d_in, const int* in_sizes, int n_in,
                              void* d_out, int out_size, void* d_ws, size_t ws_size,
                              hipStream_t stream)
{
    (void)in_sizes; (void)n_in; (void)out_size; (void)ws_size;

    const float* features  = (const float*)d_in[0];   // (8,256,128,128)
    const float* proposals = (const float*)d_in[1];   // (8,100,256)
    const float* mask_pred = (const float*)d_in[2];   // (8,100,128,128)
    auto P = [&](int i) { return (const float*)d_in[i]; };
    // params (dict insertion order after train_flag at index 3):
    const float *f_wl_w = P(4),  *f_wl_b = P(5),  *f_dn_w = P(6),  *f_dn_b = P(7);
    const float *f_ln_w = P(8),  *f_ln_b = P(9);
    const float *k_wl_w = P(10), *k_wl_b = P(11), *k_dn_w = P(12), *k_dn_b = P(13);
    const float *k_ln_w = P(14), *k_ln_b = P(15);
    const float *s_ipw = P(16), *s_ipb = P(17), *s_opw = P(18), *s_opb = P(19);
    const float *s_ln_w = P(20), *s_ln_b = P(21);
    const float *ffn_w1 = P(22), *ffn_b1 = P(23), *ffn_w2 = P(24), *ffn_b2 = P(25);
    const float *ffn_ln_w = P(26), *ffn_ln_b = P(27);
    const float *cls0_w = P(28), *cls0_ln_w = P(29), *cls0_ln_b = P(30);
    const float *mask0_w = P(31), *mask0_ln_w = P(32), *mask0_ln_b = P(33);
    const float *cls1_w = P(34), *cls1_ln_w = P(35), *cls1_ln_b = P(36);
    const float *mask1_w = P(37), *mask1_ln_w = P(38), *mask1_ln_b = P(39);
    const float *fc_cls_w = P(40), *fc_cls_b = P(41);
    const float *fc_mask_w = P(42), *fc_mask_b = P(43);

    float* out = (float*)d_out;
    float* cls_out  = out;                                 // 8*100*81   = 64800
    float* mask_out = out + 64800;                         // 8*100*16384
    float* obj_out  = out + 64800 + BATCH * NPQ * HWSZ;    // 8*100*256

    // workspace layout (floats); total 3,154,400 floats = 12.6 MB
    float* ws    = (float*)d_ws;
    float* fbuf  = ws;                       // (8,100,256)  204800  [alias: k2, x1]
    float* wbuf  = fbuf  + 204800;           // (8,100,103)   82400
    float* tmpc  = wbuf  + 82400;            // (8,100,256)  204800
    float* pwout = tmpc  + 204800;           // (8,100,256)  204800  [alias: attno, obj2]
    float* fnew  = pwout + 204800;           // (8,100,256)  204800  [alias: k3, mk]
    float* qkvb  = fnew  + 204800;           // (8,100,768)  614400
    float* h1    = qkvb  + 614400;           // (8,100,2048) 1638400

    const int rows = BATCH * NPQ;            // 800

    auto gemm = [&](const float* A, long long sAb, int lda,
                    const float* W, long long sWb, const float* bias,
                    float* Y, long long sYb, int M, int N, int K, int relu) {
        dim3 g((N + 15) / 16, (M + 15) / 16, BATCH);
        gemm_bf16_kernel<<<g, dim3(32), 0, stream>>>(A, sAb, lda, W, sWb, bias,
                                                     Y, sYb, M, N, K, relu);
    };
    auto ln = [&](const float* x, const float* resid, const float* w, const float* bb,
                  float* y, int relu) {
        ln_kernel<<<rows, HIDC, 0, stream>>>(x, resid, w, bb, y, relu);
    };

    // ---- 1. f = (sigmoid(mask)>0.5) . features   (WMMA + K-split atomics)
    zero_kernel<<<(204800 + 255) / 256, 256, 0, stream>>>(fbuf, 204800);
    pool_gemm_kernel<<<dim3(HIDC / 16, (NPQ + 15) / 16, BATCH * KSEG),
                       dim3(32), 0, stream>>>(mask_pred, features, fbuf);

    // ---- 2. dysep #1: q = proposals, v = f
    gemm(proposals, NPQ * HIDC, HIDC, f_wl_w, 0, f_wl_b, wbuf, NPQ * NW, NPQ, NW, HIDC, 0);
    dysep_conv_kernel<<<rows, HIDC, 0, stream>>>(wbuf, fbuf, tmpc);
    gemm(wbuf + 3, NPQ * NW, NW, tmpc, NPQ * HIDC, nullptr,
         pwout, NPQ * HIDC, NPQ, HIDC, NPQ, 0);
    ln(pwout, nullptr, f_dn_w, f_dn_b, tmpc, 0);        // inner LN of dysep
    ln(tmpc, fbuf, f_ln_w, f_ln_b, fnew, 0);            // f = LN(f + f_tmp)

    // ---- 3. dysep #2: q = proposals, v = f(new)
    gemm(proposals, NPQ * HIDC, HIDC, k_wl_w, 0, k_wl_b, wbuf, NPQ * NW, NPQ, NW, HIDC, 0);
    dysep_conv_kernel<<<rows, HIDC, 0, stream>>>(wbuf, fnew, tmpc);
    gemm(wbuf + 3, NPQ * NW, NW, tmpc, NPQ * HIDC, nullptr,
         pwout, NPQ * HIDC, NPQ, HIDC, NPQ, 0);
    ln(pwout, nullptr, k_dn_w, k_dn_b, tmpc, 0);
    float* k2 = fbuf;                                   // f dead -> reuse
    ln(tmpc, fnew, k_ln_w, k_ln_b, k2, 0);              // k = LN(f + f_tmp)

    // ---- 4. MHA + residual LN
    gemm(k2, NPQ * HIDC, HIDC, s_ipw, 0, s_ipb, qkvb, NPQ * 3 * HIDC, NPQ, 3 * HIDC, HIDC, 0);
    float* attno = pwout;
    mha_kernel<<<BATCH * NHEAD, 128, 0, stream>>>(qkvb, attno);
    gemm(attno, NPQ * HIDC, HIDC, s_opw, 0, s_opb, tmpc, NPQ * HIDC, NPQ, HIDC, HIDC, 0);
    float* k3 = fnew;
    ln(tmpc, k2, s_ln_w, s_ln_b, k3, 0);                // k = LN(k + mha)

    // ---- 5. FFN + residual LN  -> obj
    gemm(k3, NPQ * HIDC, HIDC, ffn_w1, 0, ffn_b1, h1, NPQ * FFC_, NPQ, FFC_, HIDC, 1);
    gemm(h1, NPQ * FFC_, FFC_, ffn_w2, 0, ffn_b2, tmpc, NPQ * HIDC, NPQ, HIDC, FFC_, 0);
    float* obj2 = pwout;
    ln(tmpc, k3, ffn_ln_w, ffn_ln_b, obj2, 0);

    // ---- 6. cls branch (train_flag == 1)
    float* x1 = fbuf;                                   // k2 dead -> reuse
    gemm(obj2, NPQ * HIDC, HIDC, cls0_w, 0, nullptr, tmpc, NPQ * HIDC, NPQ, HIDC, HIDC, 0);
    ln(tmpc, nullptr, cls0_ln_w, cls0_ln_b, x1, 1);
    gemm(x1, NPQ * HIDC, HIDC, cls1_w, 0, nullptr, tmpc, NPQ * HIDC, NPQ, HIDC, HIDC, 0);
    ln(tmpc, nullptr, cls1_ln_w, cls1_ln_b, x1, 1);
    gemm(x1, NPQ * HIDC, HIDC, fc_cls_w, 0, fc_cls_b, cls_out, NPQ * 81, NPQ, 81, HIDC, 0);

    // ---- 7. mask branch -> mask kernels
    gemm(obj2, NPQ * HIDC, HIDC, mask0_w, 0, nullptr, tmpc, NPQ * HIDC, NPQ, HIDC, HIDC, 0);
    ln(tmpc, nullptr, mask0_ln_w, mask0_ln_b, x1, 1);
    gemm(x1, NPQ * HIDC, HIDC, mask1_w, 0, nullptr, tmpc, NPQ * HIDC, NPQ, HIDC, HIDC, 0);
    ln(tmpc, nullptr, mask1_ln_w, mask1_ln_b, x1, 1);
    float* mk = fnew;                                   // k3 dead -> reuse
    gemm(x1, NPQ * HIDC, HIDC, fc_mask_w, 0, fc_mask_b, mk, NPQ * HIDC, NPQ, HIDC, HIDC, 0);

    // ---- 8. new_mask_preds = mk . features   (features L2-resident: 134MB < 192MB)
    gemm(mk, NPQ * HIDC, HIDC, features, (long long)HIDC * HWSZ, nullptr,
         mask_out, (long long)NPQ * HWSZ, NPQ, HWSZ, HIDC, 0);

    // ---- 9. obj_out = obj (8,100,256,1,1)
    copy_kernel<<<(204800 + 255) / 256, 256, 0, stream>>>(obj2, obj_out, 204800);
}
